// MyLSTM_33182917329099
// MI455X (gfx1250) — matrix operand order repaired and checked by reading, not compiled, not run on hardware
//
#include <hip/hip_runtime.h>
#include <math.h>

// Sizes from the reference
#define BB 32
#define TT 1024
#define IN 256
#define HH 512
#define NG 2048      // 4 gates * HH
#define CL 1000
#define NTC 63       // ceil(1000/16) N-tiles for output projection
#define NWG 16       // recurrence grid size

typedef __attribute__((ext_vector_type(16))) __bf16 v16bf;
typedef __attribute__((ext_vector_type(8)))  __bf16 v8bf;
typedef __attribute__((ext_vector_type(8)))  float  v8f;

union Frag { v16bf v; v8bf h[2]; };

__device__ __forceinline__ __bf16 f2bf(float f) {
  unsigned u = __builtin_bit_cast(unsigned, f);
  unsigned r = (u + 0x7FFFu + ((u >> 16) & 1u)) >> 16;
  unsigned short s = (unsigned short)r;
  return __builtin_bit_cast(__bf16, s);
}

__device__ __forceinline__ v8f wmma_bf16(const Frag& a, const Frag& b, v8f c) {
  // D = A(16x32 bf16) x B(32x16 bf16) + C(16x16 f32)
  return __builtin_amdgcn_wmma_f32_16x16x32_bf16(
      false, a.v, false, b.v, (short)0, c, false, false);
}

__device__ __forceinline__ float sigf(float x) {
  return 1.0f / (1.0f + __expf(-x));
}

// ---------------------------------------------------------------------------
// Packing kernels: f32 -> bf16 fragment-major layouts
// ---------------------------------------------------------------------------

// x [B,T,IN] f32 -> bf16 (same layout)
__global__ void pack_x(const float* __restrict__ x, __bf16* __restrict__ xbf, int n) {
  int i = blockIdx.x * blockDim.x + threadIdx.x;
  if (i < n) xbf[i] = f2bf(x[i]);
}

// W gates [IN,HH] each -> Wfrag[ntile(128)][kc(8)][lane(32)][16 bf16]
__global__ void pack_w(const float* __restrict__ Wi, const float* __restrict__ Wf,
                       const float* __restrict__ Wg, const float* __restrict__ Wo,
                       __bf16* __restrict__ wfrag) {
  int i = blockIdx.x * blockDim.x + threadIdx.x;           // < 128*8*512
  int frag = i >> 9, within = i & 511;
  int lane = within >> 4, v = within & 15;
  int ntile = frag >> 3, kc = frag & 7;
  int gate = ntile >> 5, jt = ntile & 31;
  int n = jt * 16 + (lane & 15);
  int k = kc * 32 + (lane >> 4) * 16 + v;
  const float* Ws[4] = {Wi, Wf, Wg, Wo};
  wfrag[i] = f2bf(Ws[gate][k * HH + n]);                   // B[k][n] = W[k][n]
}

// U gates [HH,HH] each, used as h @ U^T -> B[k][n] = U[n][k] (transpose)
// Ufrag[ntile(128)][kc(16)][lane(32)][16 bf16]
__global__ void pack_u(const float* __restrict__ Ui, const float* __restrict__ Uf,
                       const float* __restrict__ Ug, const float* __restrict__ Uo,
                       __bf16* __restrict__ ufrag) {
  int i = blockIdx.x * blockDim.x + threadIdx.x;           // < 128*16*512
  int frag = i >> 9, within = i & 511;
  int lane = within >> 4, v = within & 15;
  int ntile = frag >> 4, kc = frag & 15;
  int gate = ntile >> 5, jt = ntile & 31;
  int n = jt * 16 + (lane & 15);
  int k = kc * 32 + (lane >> 4) * 16 + v;
  const float* Us[4] = {Ui, Uf, Ug, Uo};
  ufrag[i] = f2bf(Us[gate][n * HH + k]);
}

// W_out [HH,CL] -> Wofrag[ntile(63)][kc(16)][lane(32)][16 bf16], zero-pad n>=1000
__global__ void pack_wo(const float* __restrict__ Wout, __bf16* __restrict__ wofrag) {
  int i = blockIdx.x * blockDim.x + threadIdx.x;           // < 63*16*512
  int frag = i >> 9, within = i & 511;
  int lane = within >> 4, v = within & 15;
  int nt = frag >> 4, kc = frag & 15;
  int n = nt * 16 + (lane & 15);
  int k = kc * 32 + (lane >> 4) * 16 + v;
  wofrag[i] = (n < CL) ? f2bf(Wout[k * CL + n]) : f2bf(0.0f);
}

// Combined gate bias [2048] + zero the grid barrier counter
__global__ void pack_misc(const float* __restrict__ bi, const float* __restrict__ bf_,
                          const float* __restrict__ bg, const float* __restrict__ bo,
                          float* __restrict__ bias, unsigned* __restrict__ cnt) {
  int i = blockIdx.x * blockDim.x + threadIdx.x;
  if (i == 0) cnt[0] = 0u;
  if (i < NG) {
    const float* bs[4] = {bi, bf_, bg, bo};
    bias[i] = bs[i >> 9][i & 511];
  }
}

// ---------------------------------------------------------------------------
// Persistent LSTM recurrence. 16 blocks x 128 threads = 64 waves = 64 units
// (2 batch tiles x 32 hidden tiles). Cell state c lives in VGPRs for all T.
// Both U (128 KB) and W (64 KB) fragment tables live in LDS (192 KB / WGP).
// An opaque zero offset regenerated per timestep stops LICM from hoisting
// the (loop-invariant) B-fragment loads out of the T loop and spilling.
// ---------------------------------------------------------------------------
__global__ __launch_bounds__(128, 1) void lstm_recur(
    const __bf16* __restrict__ xbf,
    const __bf16* wfrag,
    const __bf16* ufrag,
    const float* __restrict__ bias,
    __bf16* __restrict__ hseq,      // [B][T][HH] bf16
    float* __restrict__ outH,       // [B][HH]
    float* __restrict__ outC,       // [B][HH]
    unsigned* __restrict__ cnt, int nwg) {
  extern __shared__ char smem[];                           // 196608 bytes
  const int tid = threadIdx.x;
  const int lane = tid & 31;
  const int wv = tid >> 5;
  const int bx = blockIdx.x;

  // Stage this block's 2 j-tiles: U frags at [0,128K), W frags at [128K,192K)
  {
    const v8bf* su = (const v8bf*)ufrag;
    const v8bf* sw = (const v8bf*)wfrag;
    v8bf* dst = (v8bf*)smem;
    for (int idx = tid; idx < 12288; idx += 128) {
      if (idx < 8192) {
        int frag = idx >> 6, within = idx & 63;
        int jl = frag >> 6, rem = frag & 63;
        int gate = rem >> 4, kc = rem & 15;
        int gf = (gate * 32 + (bx * 2 + jl)) * 16 + kc;
        dst[idx] = su[(size_t)gf * 64 + within];
      } else {
        int i2 = idx - 8192;
        int frag = i2 >> 6, within = i2 & 63;
        int jl = frag >> 5, rem = frag & 31;
        int gate = rem >> 3, kc = rem & 7;
        int gf = (gate * 32 + (bx * 2 + jl)) * 8 + kc;
        dst[idx] = sw[(size_t)gf * 64 + within];
      }
    }
  }
  __syncthreads();

  const int unit = bx * 4 + wv;               // 0..63
  const int mt = unit & 1, jt = unit >> 1, jloc = wv >> 1;
  const int m0 = mt * 16, j0 = jt * 16;
  const int rowA = m0 + (lane & 15);          // batch row for A fragments
  const int colShift = (lane >> 4) * 16;      // A-layout K shift for hi lanes
  const int bRow = m0 + ((lane >> 4) << 3);   // C-layout batch base (add r)
  const int jCol = j0 + (lane & 15);          // C-layout hidden column

  const __bf16* xA = xbf + (size_t)rowA * TT * IN + colShift;
  const __bf16* hA = hseq + (size_t)rowA * TT * HH + colShift;

  float bv[4];
#pragma unroll
  for (int g = 0; g < 4; ++g) bv[g] = bias[g * HH + jCol];

  float cst[8];
#pragma unroll
  for (int r = 0; r < 8; ++r) cst[r] = 0.0f;

  for (int t = 0; t < TT; ++t) {
    // Opaque zero: makes LDS fragment addresses non-invariant across t
    unsigned sb = 0;
    asm volatile("" : "+v"(sb));
    const char* uB = smem + sb + jloc * 65536 + lane * 32;
    const char* wB = smem + sb + 131072 + jloc * 32768 + lane * 32;

    v8f acc[4] = {};   // zero-init -> first WMMA uses inline SRC2 = 0

    // Gate pre-activations: x_t @ W (K = 256), W frags from LDS
    const __bf16* xAt = xA + (size_t)t * IN;
#pragma unroll 4
    for (int kc = 0; kc < 8; ++kc) {
      Frag a;
      a.h[0] = *(const v8bf*)(xAt + kc * 32);
      a.h[1] = *(const v8bf*)(xAt + kc * 32 + 8);
#pragma unroll
      for (int g = 0; g < 4; ++g) {
        const char* bp = wB + (g * 8 + kc) * 1024;
        Frag b;
        b.h[0] = *(const v8bf*)(bp);
        b.h[1] = *(const v8bf*)(bp + 16);
        acc[g] = wmma_bf16(a, b, acc[g]);
      }
    }

    // + h_{t-1} @ U^T (K = 512), U frags from LDS
    if (t > 0) {
      const __bf16* hAt = hA + (size_t)(t - 1) * HH;
#pragma unroll 4
      for (int kc = 0; kc < 16; ++kc) {
        Frag a;
        a.h[0] = *(const v8bf*)(hAt + kc * 32);
        a.h[1] = *(const v8bf*)(hAt + kc * 32 + 8);
#pragma unroll
        for (int g = 0; g < 4; ++g) {
          const char* bp = uB + (g * 16 + kc) * 1024;
          Frag b;
          b.h[0] = *(const v8bf*)(bp);
          b.h[1] = *(const v8bf*)(bp + 16);
          acc[g] = wmma_bf16(a, b, acc[g]);
        }
      }
    }

    // Elementwise cell update; bias folded in here; c stays in registers
#pragma unroll
    for (int r = 0; r < 8; ++r) {
      float iv = sigf(acc[0][r] + bv[0]);
      float fv = sigf(acc[1][r] + bv[1]);
      float gv = tanhf(acc[2][r] + bv[2]);
      float ov = sigf(acc[3][r] + bv[3]);
      float cc = fv * cst[r] + iv * gv;
      cst[r] = cc;
      float hv = ov * tanhf(cc);
      int b = bRow + r;
      hseq[((size_t)b * TT + t) * HH + jCol] = f2bf(hv);
      if (t == TT - 1) outH[b * HH + jCol] = hv;
    }

    if (t + 1 < TT) __builtin_prefetch(xA + (size_t)(t + 1) * IN, 0, 0);

    // Grid-wide phase barrier (monotone counter; re-zeroed by pack_misc)
    __syncthreads();
    if (tid == 0) {
      __threadfence();
      __hip_atomic_fetch_add(cnt, 1u, __ATOMIC_RELEASE, __HIP_MEMORY_SCOPE_AGENT);
      unsigned tgt = (unsigned)(t + 1) * (unsigned)nwg;
      while (__hip_atomic_load(cnt, __ATOMIC_ACQUIRE, __HIP_MEMORY_SCOPE_AGENT) < tgt)
        __builtin_amdgcn_s_sleep(1);
      __threadfence();
    }
    __syncthreads();
  }

#pragma unroll
  for (int r = 0; r < 8; ++r) outC[(bRow + r) * HH + jCol] = cst[r];
}

// ---------------------------------------------------------------------------
// logits = hseq @ W_out + b_out   (M=32768, N=1008 padded, K=512), WMMA.
// A tile staged in LDS; each wave keeps all 16 A fragments in VGPRs across
// its 8 N-tiles; accumulators start at inline 0, bias folded into the store.
// ---------------------------------------------------------------------------
__global__ void logits_gemm(const __bf16* __restrict__ hseq,
                            const __bf16* __restrict__ wofrag,
                            const float* __restrict__ bout,
                            float* __restrict__ out) {
  __shared__ __bf16 atile[16 * HH];                        // 16 KB A tile
  const int tid = threadIdx.x, lane = tid & 31, wv = tid >> 5;
  const int r0 = blockIdx.x * 16;                          // 16 (b,t) rows
  {
    const v8bf* src = (const v8bf*)(hseq + (size_t)r0 * HH);
    v8bf* dst = (v8bf*)atile;
    for (int i = tid; i < 1024; i += 256) dst[i] = src[i];
  }
  __syncthreads();

  const int nlo = lane & 15, nhi = lane >> 4;
  const __bf16* aBase = atile + nlo * HH + nhi * 16;

  // Hold the full-K A fragments (16 x 8 VGPRs) in registers
  Frag afr[16];
#pragma unroll
  for (int kc = 0; kc < 16; ++kc) {
    afr[kc].h[0] = *(const v8bf*)(aBase + kc * 32);
    afr[kc].h[1] = *(const v8bf*)(aBase + kc * 32 + 8);
  }

  for (int nt = wv; nt < NTC; nt += 8) {
    int col = nt * 16 + nlo;
    v8f acc = {};
#pragma unroll 4
    for (int kc = 0; kc < 16; ++kc) {
      const __bf16* bp = wofrag + ((size_t)(nt * 16 + kc)) * 512 + lane * 16;
      Frag b;
      b.h[0] = *(const v8bf*)(bp);
      b.h[1] = *(const v8bf*)(bp + 8);
      acc = wmma_bf16(afr[kc], b, acc);
    }
    if (col < CL) {
      float bb = bout[col];
#pragma unroll
      for (int r = 0; r < 8; ++r) {
        int row = r0 + r + (nhi << 3);
        out[(size_t)row * CL + col] = acc[r] + bb;
      }
    }
  }
}

// ---------------------------------------------------------------------------
// In-place log_softmax over last dim (1000), one wave32 per row
// ---------------------------------------------------------------------------
__global__ void logsoftmax_k(float* __restrict__ out) {
  const int lane = threadIdx.x & 31, wv = threadIdx.x >> 5;
  const size_t row = (size_t)blockIdx.x * 8 + wv;
  float* p = out + row * CL;
  float v[32];
  float mx = -1e30f;
#pragma unroll
  for (int i = 0; i < 32; ++i) {
    int idx = lane + i * 32;
    v[i] = (idx < CL) ? p[idx] : -1e30f;
    mx = fmaxf(mx, v[i]);
  }
  for (int m = 16; m; m >>= 1) mx = fmaxf(mx, __shfl_xor(mx, m, 32));
  float s = 0.0f;
#pragma unroll
  for (int i = 0; i < 32; ++i) s += __expf(v[i] - mx);
  for (int m = 16; m; m >>= 1) s += __shfl_xor(s, m, 32);
  float lse = __logf(s) + mx;
#pragma unroll
  for (int i = 0; i < 32; ++i) {
    int idx = lane + i * 32;
    if (idx < CL) p[idx] = v[i] - lse;
  }
}

// ---------------------------------------------------------------------------
// Launch
// ---------------------------------------------------------------------------
extern "C" void kernel_launch(void* const* d_in, const int* in_sizes, int n_in,
                              void* d_out, int out_size, void* d_ws, size_t ws_size,
                              hipStream_t stream) {
  (void)in_sizes; (void)n_in; (void)out_size; (void)ws_size;
  const float* x    = (const float*)d_in[0];
  const float* W_i  = (const float*)d_in[1];
  const float* U_i  = (const float*)d_in[2];
  const float* b_i  = (const float*)d_in[3];
  const float* W_f  = (const float*)d_in[4];
  const float* U_f  = (const float*)d_in[5];
  const float* b_f  = (const float*)d_in[6];
  const float* W_g  = (const float*)d_in[7];
  const float* U_g  = (const float*)d_in[8];
  const float* b_g  = (const float*)d_in[9];
  const float* W_o  = (const float*)d_in[10];
  const float* U_o  = (const float*)d_in[11];
  const float* b_o  = (const float*)d_in[12];
  const float* Wout = (const float*)d_in[13];
  const float* bout = (const float*)d_in[14];
  float* out = (float*)d_out;

  // Workspace carve-up (~55 MB total)
  char* w = (char*)d_ws;
  size_t off = 0;
  auto carve = [&](size_t bytes) {
    void* p = w + off;
    off = (off + bytes + 255) & ~(size_t)255;
    return p;
  };
  __bf16* xbf    = (__bf16*)carve((size_t)BB * TT * IN * 2);        // 16 MB
  __bf16* wfrag  = (__bf16*)carve((size_t)128 * 8 * 512 * 2);       // 1 MB
  __bf16* ufrag  = (__bf16*)carve((size_t)128 * 16 * 512 * 2);      // 2 MB
  __bf16* wofrag = (__bf16*)carve((size_t)NTC * 16 * 512 * 2);      // ~1 MB
  float*  bias   = (float*)carve((size_t)NG * 4);                   // 8 KB
  __bf16* hseq   = (__bf16*)carve((size_t)BB * TT * HH * 2);        // 32 MB
  unsigned* cnt  = (unsigned*)carve(256);

  float* outLogits = out;                                  // [B*T, 1000]
  float* outH = out + (size_t)BB * TT * CL;                // [1,B,H]
  float* outC = outH + (size_t)BB * HH;                    // [1,B,H]

  // 1) Pack weights / inputs to bf16 fragment layouts
  pack_x   <<<(BB * TT * IN) / 256, 256, 0, stream>>>(x, xbf, BB * TT * IN);
  pack_w   <<<(128 * 8 * 512) / 256, 256, 0, stream>>>(W_i, W_f, W_g, W_o, wfrag);
  pack_u   <<<(128 * 16 * 512) / 256, 256, 0, stream>>>(U_i, U_f, U_g, U_o, ufrag);
  pack_wo  <<<(NTC * 16 * 512) / 256, 256, 0, stream>>>(Wout, wofrag);
  pack_misc<<<8, 256, 0, stream>>>(b_i, b_f, b_g, b_o, bias, cnt);

  // 2) Persistent recurrence: 16 blocks, 192 KB LDS each (one WGP per block)
  lstm_recur<<<NWG, 128, 196608, stream>>>(xbf, wfrag, ufrag, bias, hseq,
                                           outH, outC, cnt, NWG);

  // 3) Output projection + log_softmax
  logits_gemm<<<(BB * TT) / 16, 256, 0, stream>>>(hseq, wofrag, bout, outLogits);
  logsoftmax_k<<<(BB * TT) / 8, 256, 0, stream>>>(outLogits);
}
